// GatedAttnPool_84894323573139
// MI455X (gfx1250) — compile-verified, matrix operand order checked by reading, not run on hardware
//
#include <hip/hip_runtime.h>
#include <hip/hip_bf16.h>

typedef _Float16 v8h  __attribute__((ext_vector_type(8)));
typedef _Float16 v16h __attribute__((ext_vector_type(16)));
typedef float    v8f  __attribute__((ext_vector_type(8)));

#define D_DIM 512
#define H_DIM 128
#define ROWS_PER_BLOCK 32
#define LDS_K (D_DIM + 8)   // padded LDS row stride (halves) -> conflict-free ds_load_b128

// ---------------------------------------------------------------------------
// Kernel 0: W1 [512,128] f32 row-major  ->  W1t [128,512] f16 row-major
// (gate-column-major so WMMA B-fragments are contiguous 16B chunks along K)
// ---------------------------------------------------------------------------
__global__ void k_w1_to_f16t(const float* __restrict__ W1, _Float16* __restrict__ W1t) {
    int idx = blockIdx.x * blockDim.x + threadIdx.x;
    if (idx >= D_DIM * H_DIM) return;
    int k = idx >> 7;           // idx / 128
    int n = idx & (H_DIM - 1);  // idx % 128
    W1t[n * D_DIM + k] = (_Float16)W1[idx];
}

// ---------------------------------------------------------------------------
// Kernel 1: per-node gate logit = tanh(h@W1 + b1) @ W2 + b2   via WMMA f16
// 256 threads = 8 waves; 32 rows of h per block; wave w owns H-columns [16w,16w+16)
// ---------------------------------------------------------------------------
__global__ __launch_bounds__(256)
void k_gate_logits(const float* __restrict__ h,
                   const _Float16* __restrict__ W1t,
                   const float* __restrict__ b1,
                   const float* __restrict__ W2,
                   const float* __restrict__ b2,
                   float* __restrict__ logits,
                   int N)
{
    __shared__ __align__(16) _Float16 shA[ROWS_PER_BLOCK][LDS_K];
    __shared__ float shLogit[ROWS_PER_BLOCK];

    const int tid  = threadIdx.x;
    const int row0 = blockIdx.x * ROWS_PER_BLOCK;

    if (tid < ROWS_PER_BLOCK) shLogit[tid] = 0.f;

    // cooperative load + f32->f16 convert of the 32x512 tile (float4-coalesced)
    for (int i = tid; i < ROWS_PER_BLOCK * (D_DIM / 4); i += 256) {
        int r  = i >> 7;        // i / 128 float4's per row
        int c4 = i & 127;
        float4 v = make_float4(0.f, 0.f, 0.f, 0.f);
        if (row0 + r < N)
            v = *(const float4*)&h[(size_t)(row0 + r) * D_DIM + c4 * 4];
        _Float16* dst = &shA[r][c4 * 4];
        dst[0] = (_Float16)v.x; dst[1] = (_Float16)v.y;
        dst[2] = (_Float16)v.z; dst[3] = (_Float16)v.w;
    }
    __syncthreads();

    const int wave = tid >> 5;   // 0..7 -> H-column tile
    const int lane = tid & 31;
    const int half = lane >> 4;  // K-half selector per ISA A/B f16 layout
    const int l16  = lane & 15;

    const int   col = wave * 16 + l16;   // gate column this lane owns in C-frag
    const float b1c = b1[col];
    const float w2c = W2[col];

    v8f acc0 = {};   // rows 0..15 of tile
    v8f acc1 = {};   // rows 16..31 of tile

    const _Float16* bRow = &W1t[(size_t)col * D_DIM];

    #pragma unroll
    for (int kb = 0; kb < D_DIM; kb += 32) {
        // B fragment (32x16, K-major within lane): two contiguous 16B chunks
        v8h b_lo = *(const v8h*)&bRow[kb + half * 8];
        v8h b_hi = *(const v8h*)&bRow[kb + 16 + half * 8];
        v16h bF = __builtin_shufflevector(b_lo, b_hi,
                    0,1,2,3,4,5,6,7,8,9,10,11,12,13,14,15);

        // A fragments (16x32): lane holds row l16, K per half
        v8h a0_lo = *(const v8h*)&shA[l16][kb + half * 8];
        v8h a0_hi = *(const v8h*)&shA[l16][kb + 16 + half * 8];
        v16h aF0 = __builtin_shufflevector(a0_lo, a0_hi,
                    0,1,2,3,4,5,6,7,8,9,10,11,12,13,14,15);

        v8h a1_lo = *(const v8h*)&shA[16 + l16][kb + half * 8];
        v8h a1_hi = *(const v8h*)&shA[16 + l16][kb + 16 + half * 8];
        v16h aF1 = __builtin_shufflevector(a1_lo, a1_hi,
                    0,1,2,3,4,5,6,7,8,9,10,11,12,13,14,15);

        acc0 = __builtin_amdgcn_wmma_f32_16x16x32_f16(false, aF0, false, bF,
                                                      (short)0, acc0, false, false);
        acc1 = __builtin_amdgcn_wmma_f32_16x16x32_f16(false, aF1, false, bF,
                                                      (short)0, acc1, false, false);
    }

    // C layout: VGPR v -> row v (lanes 0-15) / row v+8 (lanes 16-31), N = l16.
    // tanh + W2 weighting, then xor-shuffle reduce across the 16 columns.
    #pragma unroll
    for (int v = 0; v < 8; ++v) {
        float s0 = tanhf(acc0[v] + b1c) * w2c;
        float s1 = tanhf(acc1[v] + b1c) * w2c;
        #pragma unroll
        for (int m = 1; m < 16; m <<= 1) {
            s0 += __shfl_xor(s0, m, 32);
            s1 += __shfl_xor(s1, m, 32);
        }
        if (l16 == 0) {
            atomicAdd(&shLogit[v + 8 * half], s0);          // rows 0..15
            atomicAdd(&shLogit[16 + v + 8 * half], s1);     // rows 16..31
        }
    }
    __syncthreads();

    if (tid < ROWS_PER_BLOCK && row0 + tid < N)
        logits[row0 + tid] = shLogit[tid] + b2[0];
}

// ---------------------------------------------------------------------------
// Kernel 2: per-graph segment softmax + weighted pooling.
// One block per graph; binary-search segment bounds in sorted batch[].
// ---------------------------------------------------------------------------
__global__ __launch_bounds__(256)
void k_segment_pool(const float* __restrict__ h,
                    const int* __restrict__ batch,
                    const float* __restrict__ logits,
                    float* __restrict__ pooled,
                    int N)
{
    __shared__ float red[256];
    __shared__ int   seg[2];
    __shared__ float stat[2];
    __shared__ float a_sh[256];

    const int g   = blockIdx.x;
    const int tid = threadIdx.x;

    if (tid == 0) {
        int lo = 0, hi = N;
        while (lo < hi) { int mid = (lo + hi) >> 1; if (batch[mid] < g) lo = mid + 1; else hi = mid; }
        seg[0] = lo;
        hi = N;
        while (lo < hi) { int mid = (lo + hi) >> 1; if (batch[mid] < g + 1) lo = mid + 1; else hi = mid; }
        seg[1] = lo;
    }
    __syncthreads();
    const int start = seg[0], end = seg[1];

    // ---- segment max ----
    float m = -3.402823466e+38f;
    for (int i = start + tid; i < end; i += 256) m = fmaxf(m, logits[i]);
    red[tid] = m;
    __syncthreads();
    for (int s = 128; s > 0; s >>= 1) {
        if (tid < s) red[tid] = fmaxf(red[tid], red[tid + s]);
        __syncthreads();
    }
    if (tid == 0) stat[0] = red[0];
    __syncthreads();
    m = stat[0];

    // ---- segment sum of exp ----
    float sum = 0.f;
    for (int i = start + tid; i < end; i += 256) sum += __expf(logits[i] - m);
    red[tid] = sum;
    __syncthreads();
    for (int s = 128; s > 0; s >>= 1) {
        if (tid < s) red[tid] += red[tid + s];
        __syncthreads();
    }
    if (tid == 0) stat[1] = (red[0] > 0.f) ? (1.f / red[0]) : 0.f;
    __syncthreads();
    const float inv_denom = stat[1];

    // ---- weighted pooling: each thread owns 2 of the 512 columns ----
    float2 acc = make_float2(0.f, 0.f);
    for (int base = start; base < end; base += 256) {
        int cnt = min(256, end - base);
        if (tid < cnt) a_sh[tid] = __expf(logits[base + tid] - m) * inv_denom;
        __syncthreads();
        #pragma unroll 4
        for (int j = 0; j < cnt; ++j) {
            float a = a_sh[j];
            float2 hv = *(const float2*)&h[(size_t)(base + j) * D_DIM + tid * 2];
            acc.x += hv.x * a;
            acc.y += hv.y * a;
        }
        __syncthreads();
    }
    *(float2*)&pooled[(size_t)g * D_DIM + tid * 2] = acc;
}

// ---------------------------------------------------------------------------
extern "C" void kernel_launch(void* const* d_in, const int* in_sizes, int n_in,
                              void* d_out, int out_size, void* d_ws, size_t ws_size,
                              hipStream_t stream) {
    const float* h     = (const float*)d_in[0];
    const int*   batch = (const int*)d_in[1];
    const float* W1    = (const float*)d_in[2];
    const float* b1    = (const float*)d_in[3];
    const float* W2    = (const float*)d_in[4];
    const float* b2    = (const float*)d_in[5];
    float* pooled = (float*)d_out;

    const int N = in_sizes[0] / D_DIM;
    const int G = out_size / D_DIM;

    float* logits = (float*)d_ws;
    size_t w1t_off = (((size_t)N * sizeof(float)) + 255) & ~(size_t)255;
    _Float16* W1t = (_Float16*)((char*)d_ws + w1t_off);

    // Kernel 0: convert + transpose W1
    k_w1_to_f16t<<<(D_DIM * H_DIM + 255) / 256, 256, 0, stream>>>(W1, W1t);

    // Kernel 1: WMMA gate logits
    int nblk = (N + ROWS_PER_BLOCK - 1) / ROWS_PER_BLOCK;
    k_gate_logits<<<nblk, 256, 0, stream>>>(h, W1t, b1, W2, b2, logits, N);

    // Kernel 2: segment softmax + pooling
    k_segment_pool<<<G, 256, 0, stream>>>(h, batch, logits, pooled, N);
}
